// GATAggregator_global_38190849196689
// MI455X (gfx1250) — matrix operand order repaired and checked by reading, not compiled, not run on hardware
//
#include <hip/hip_runtime.h>
#include <hip/hip_bf16.h>
#include <math.h>

// ---------------------------------------------------------------------------
// GAT aggregator for MI455X (gfx1250, wave32).
//  - Dense feature GEMMs ([N,64]x[64,192]) on the matrix cores:
//    B (24 KB) is staged once per workgroup into LDS pre-swizzled into WMMA
//    fragment order; each wave keeps its A fragments resident and sweeps all
//    12 column tiles -> 24 v_wmma_f32_16x16x32_bf16 per wave, B fragments
//    come from aligned ds_load_b128.
//  - Irregular edge phase (edge softmax + scatter aggregation) uses float
//    atomics; feat/agg (76.8 MB each) fit in the 192 MB L2 so the random
//    per-edge traffic stays on-die.
// ---------------------------------------------------------------------------

typedef __attribute__((ext_vector_type(16))) __bf16 v16bf;
typedef __attribute__((ext_vector_type(8)))  float  v8f;

#define HDIM  64
#define NHEAD 3
#define FDIM  192                 // NHEAD * HDIM
#define NTILES (FDIM / 16)        // 12
#define BFRAG_ELEMS (NTILES * 2 * 32 * 16)   // 12288 == 64*192 (permutation of B)

// ---- float atomic max via ordered-int trick (works for mixed signs) -------
__device__ __forceinline__ void atomicMaxF(float* addr, float val) {
  if (val >= 0.0f) atomicMax((int*)addr, __float_as_int(val));
  else             atomicMin((unsigned int*)addr, __float_as_uint(val));
}

// ---------------------------------------------------------------------------
__global__ void fill_f32_kernel(float* __restrict__ p, float v, int n) {
  int i = blockIdx.x * blockDim.x + threadIdx.x;
  if (i < n) p[i] = v;
}

__global__ void cvt_f32_to_bf16_kernel(const float* __restrict__ s,
                                       __bf16* __restrict__ d, int n) {
  int i = blockIdx.x * blockDim.x + threadIdx.x;
  if (i < n) d[i] = (__bf16)s[i];
}

// row = 16 x float4 ; vectorized gather
__global__ void embed_gather_kernel(const float4* __restrict__ Wemb4,
                                    const int* __restrict__ ids,
                                    float4* __restrict__ out4, int N) {
  int i = blockIdx.x * blockDim.x + threadIdx.x;
  if (i >= N * (HDIM / 4)) return;
  int n = i >> 4, c = i & 15;
  out4[i] = Wemb4[ids[n] * (HDIM / 4) + c];
}

// ---------------------------------------------------------------------------
// GEMM: C[M,192] = A[M,64] (bf16) * B[64,192] (bf16), f32 accumulate.
// One wave owns one 16-row strip and sweeps all 12 column tiles.
// Fragment layouts as used here (wave32; A/B in contiguous k order, D per ISA):
//   A 16x32 bf16 : lane L holds row m=L&15; VGPR pair p -> K = 2p + 16*(L>>4)
//   B 32x16 bf16 : lane L holds col n=L&15; element e -> K = e + 16*(L>>4)
//   D 16x16 f32  : VGPR r -> row r + 8*(L>>4), col = L&15
// B is staged in LDS already permuted to fragment order:
//   lds[(((t*2 + c)*32 + L)*16 + e]  <-  B[(c*32 + (L>>4)*16 + e)*192 + t*16 + (L&15)]
// so each lane reads its 16-element fragment as two aligned b128 DS loads.
// ---------------------------------------------------------------------------
__global__ __launch_bounds__(256)
void gemm_bf16_wmma_kernel(const __bf16* __restrict__ A,
                           const __bf16* __restrict__ B,
                           float* __restrict__ C, int M) {
  __shared__ __bf16 ldsB[BFRAG_ELEMS];      // 24576 bytes

  const int tid = threadIdx.x;

  // Cooperative fill: fragment-order permutation of B.
  for (int f = tid; f < BFRAG_ELEMS; f += 256) {
    int e  = f & 15;
    int L  = (f >> 4) & 31;
    int tc = f >> 9;                        // t*2 + c, 0..23
    int c  = tc & 1;
    int t  = tc >> 1;
    int k  = c * 32 + (L >> 4) * 16 + e;
    int n  = t * 16 + (L & 15);
    ldsB[f] = B[k * FDIM + n];
  }
  __syncthreads();

  const int wid  = (blockIdx.x * blockDim.x + tid) >> 5;  // wave id == tileM
  const int lane = tid & 31;
  if (wid * 16 >= M) return;                // wave-uniform: EXEC stays all-1

  const int m  = lane & 15;
  const int hp = lane >> 4;

  // A fragments (resident for all 12 column tiles).
  const __bf16* arow = A + (wid * 16 + m) * HDIM;
  v16bf a0, a1;
#pragma unroll
  for (int p = 0; p < 8; ++p) {
    int kb = 16 * hp + 2 * p;
    a0[2 * p]     = arow[kb];
    a0[2 * p + 1] = arow[kb + 1];
    a1[2 * p]     = arow[32 + kb];
    a1[2 * p + 1] = arow[32 + kb + 1];
  }

  float* crow = C + (size_t)(wid * 16) * FDIM;

#pragma unroll
  for (int t = 0; t < NTILES; ++t) {
    const v16bf b0 = *(const v16bf*)&ldsB[((t * 2 + 0) * 32 + lane) * 16];
    const v16bf b1 = *(const v16bf*)&ldsB[((t * 2 + 1) * 32 + lane) * 16];

    v8f acc = {};
    acc = __builtin_amdgcn_wmma_f32_16x16x32_bf16(false, a0, false, b0,
                                                  (short)0, acc, false, false);
    acc = __builtin_amdgcn_wmma_f32_16x16x32_bf16(false, a1, false, b1,
                                                  (short)0, acc, false, false);

    // M is a multiple of 16 here, so the whole 16-row strip is in range:
    // unpredicated stores (no per-row exec juggling).
#pragma unroll
    for (int r = 0; r < 8; ++r)
      crow[(r + 8 * hp) * FDIM + t * 16 + m] = acc[r];
  }
}

// ---------------------------------------------------------------------------
// el[n,h] = <feat[n,h,:], al[h,:]> ; er likewise. One thread per (n,h).
__global__ void attn_coeff_kernel(const float* __restrict__ feat,
                                  const float* __restrict__ al,
                                  const float* __restrict__ ar,
                                  float* __restrict__ el,
                                  float* __restrict__ er, int N) {
  int i = blockIdx.x * blockDim.x + threadIdx.x;
  if (i >= N * NHEAD) return;
  int n = i / NHEAD, h = i % NHEAD;
  const float* f = feat + n * FDIM + h * HDIM;
  const float* a = al + h * HDIM;
  const float* r = ar + h * HDIM;
  float sl = 0.f, sr = 0.f;
#pragma unroll 8
  for (int d = 0; d < HDIM; ++d) { sl += f[d] * a[d]; sr += f[d] * r[d]; }
  el[i] = sl;
  er[i] = sr;
}

// e = leaky_relu(el[src]+er[dst]); segment max over dst via float atomics.
__global__ void edge_attn_kernel(const float* __restrict__ el,
                                 const float* __restrict__ er,
                                 const int* __restrict__ src,
                                 const int* __restrict__ dst,
                                 float* __restrict__ ee,
                                 float* __restrict__ emax, int E) {
  int i = blockIdx.x * blockDim.x + threadIdx.x;
  if (i >= E * NHEAD) return;
  int e = i / NHEAD, h = i % NHEAD;
  float v = el[src[e] * NHEAD + h] + er[dst[e] * NHEAD + h];
  v = (v > 0.f) ? v : 0.2f * v;
  ee[i] = v;
  atomicMaxF(&emax[dst[e] * NHEAD + h], v);
}

// w = exp(e - emax[dst]); denom[dst] += w; agg[dst,h,:] += w*feat[src,h,:]
__global__ void edge_msg_kernel(const float* __restrict__ ee,
                                const float* __restrict__ feat,
                                const int* __restrict__ src,
                                const int* __restrict__ dst,
                                const float* __restrict__ emax,
                                float* __restrict__ denom,
                                float* __restrict__ agg, int E) {
  int i = blockIdx.x * blockDim.x + threadIdx.x;
  if (i >= E * NHEAD) return;
  int e = i / NHEAD, h = i % NHEAD;
  int s = src[e], t = dst[e];
  float em = emax[t * NHEAD + h];
  if (!isfinite(em)) em = 0.f;
  float w = __expf(ee[i] - em);
  atomicAdd(&denom[t * NHEAD + h], w);
  const float* f = feat + s * FDIM + h * HDIM;
  float*       a = agg  + t * FDIM + h * HDIM;
#pragma unroll 4
  for (int d = 0; d < HDIM; ++d) atomicAdd(&a[d], w * f[d]);
}

// out[n,d] = mean_h( agg[n,h,d]/max(denom[n,h],1e-9) + b[h,d] )
__global__ void finalize_kernel(const float* __restrict__ agg,
                                const float* __restrict__ denom,
                                const float* __restrict__ b,
                                float* __restrict__ out, int N) {
  int i = blockIdx.x * blockDim.x + threadIdx.x;
  if (i >= N * HDIM) return;
  int n = i >> 6, d = i & 63;
  float s = 0.f;
#pragma unroll
  for (int h = 0; h < NHEAD; ++h)
    s += agg[n * FDIM + h * HDIM + d] / fmaxf(denom[n * NHEAD + h], 1e-9f)
       + b[h * HDIM + d];
  out[i] = s * (1.0f / 3.0f);
}

// per-graph max pooling
__global__ void pool_max_kernel(const float* __restrict__ x,
                                const int* __restrict__ gid,
                                float* __restrict__ out, int N) {
  int i = blockIdx.x * blockDim.x + threadIdx.x;
  if (i >= N * HDIM) return;
  int n = i >> 6, d = i & 63;
  atomicMaxF(&out[gid[n] * HDIM + d], x[i]);
}

// ---------------------------------------------------------------------------
static void run_gat_layer(const float* x, const float* W, const float* al,
                          const float* ar, const float* b, const int* src,
                          const int* dst, int N, int E, __bf16* Abf,
                          __bf16* Wbf, float* feat, float* el, float* er,
                          float* ee, float* emax, float* denom, float* agg,
                          float* out, hipStream_t stream) {
  const int T = 256;
  cvt_f32_to_bf16_kernel<<<(N * HDIM + T - 1) / T, T, 0, stream>>>(x, Abf, N * HDIM);
  cvt_f32_to_bf16_kernel<<<(HDIM * FDIM + T - 1) / T, T, 0, stream>>>(W, Wbf, HDIM * FDIM);

  int waves  = (N + 15) / 16;                // one wave per 16-row strip
  int blocks = (waves + 7) / 8;              // 8 waves (256 threads) per block
  gemm_bf16_wmma_kernel<<<blocks, T, 0, stream>>>(Abf, Wbf, feat, N);

  attn_coeff_kernel<<<(N * NHEAD + T - 1) / T, T, 0, stream>>>(feat, al, ar, el, er, N);

  fill_f32_kernel<<<(N * NHEAD + T - 1) / T, T, 0, stream>>>(emax, -INFINITY, N * NHEAD);
  fill_f32_kernel<<<(N * NHEAD + T - 1) / T, T, 0, stream>>>(denom, 0.f, N * NHEAD);
  fill_f32_kernel<<<(N * FDIM + T - 1) / T, T, 0, stream>>>(agg, 0.f, N * FDIM);

  edge_attn_kernel<<<(E * NHEAD + T - 1) / T, T, 0, stream>>>(el, er, src, dst, ee, emax, E);
  edge_msg_kernel<<<(E * NHEAD + T - 1) / T, T, 0, stream>>>(ee, feat, src, dst, emax, denom, agg, E);

  finalize_kernel<<<(N * HDIM + T - 1) / T, T, 0, stream>>>(agg, denom, b, out, N);
}

extern "C" void kernel_launch(void* const* d_in, const int* in_sizes, int n_in,
                              void* d_out, int out_size, void* d_ws, size_t ws_size,
                              hipStream_t stream) {
  const int*   node_ids      = (const int*)d_in[0];
  const int*   edge_src      = (const int*)d_in[1];
  const int*   edge_dst      = (const int*)d_in[2];
  const int*   node_graph_id = (const int*)d_in[3];
  const float* W_embed       = (const float*)d_in[4];
  const float* W1  = (const float*)d_in[5];
  const float* al1 = (const float*)d_in[6];
  const float* ar1 = (const float*)d_in[7];
  const float* b1  = (const float*)d_in[8];
  const float* W2  = (const float*)d_in[9];
  const float* al2 = (const float*)d_in[10];
  const float* ar2 = (const float*)d_in[11];
  const float* b2  = (const float*)d_in[12];

  const int N = in_sizes[0];
  const int E = in_sizes[1];

  // Workspace carve-up (256-byte aligned slabs).
  char* base = (char*)d_ws;
  size_t off = 0;
  auto alloc = [&](size_t bytes) -> void* {
    void* p = base + off;
    off += (bytes + 255) & ~(size_t)255;
    return p;
  };
  float*  h_cur = (float*)alloc((size_t)N * HDIM * 4);
  float*  h_nxt = (float*)alloc((size_t)N * HDIM * 4);
  __bf16* Abf   = (__bf16*)alloc((size_t)N * HDIM * 2);
  __bf16* Wbf   = (__bf16*)alloc((size_t)HDIM * FDIM * 2);
  float*  feat  = (float*)alloc((size_t)N * FDIM * 4);
  float*  el    = (float*)alloc((size_t)N * NHEAD * 4);
  float*  er    = (float*)alloc((size_t)N * NHEAD * 4);
  float*  ee    = (float*)alloc((size_t)E * NHEAD * 4);
  float*  emax  = (float*)alloc((size_t)N * NHEAD * 4);
  float*  denom = (float*)alloc((size_t)N * NHEAD * 4);
  float*  agg   = (float*)alloc((size_t)N * FDIM * 4);
  (void)ws_size; (void)n_in;

  const int T = 256;

  // 1) embedding gather (vectorized: 16 float4 per row)
  embed_gather_kernel<<<(N * (HDIM / 4) + T - 1) / T, T, 0, stream>>>(
      (const float4*)W_embed, node_ids, (float4*)h_cur, N);

  // 2) two GAT layers
  run_gat_layer(h_cur, W1, al1, ar1, b1, edge_src, edge_dst, N, E,
                Abf, Wbf, feat, el, er, ee, emax, denom, agg, h_nxt, stream);
  run_gat_layer(h_nxt, W2, al2, ar2, b2, edge_src, edge_dst, N, E,
                Abf, Wbf, feat, el, er, ee, emax, denom, agg, h_cur, stream);

  // 3) per-graph max pooling into d_out [10, 64]
  float* out = (float*)d_out;
  fill_f32_kernel<<<(out_size + T - 1) / T, T, 0, stream>>>(out, -INFINITY, out_size);
  pool_max_kernel<<<(N * HDIM + T - 1) / T, T, 0, stream>>>(h_cur, node_graph_id, out, N);
}
